// Attention_850403524681
// MI455X (gfx1250) — compile-verified, matrix operand order checked by reading, not run on hardware
//
#include <hip/hip_runtime.h>

// ---------------------------------------------------------------------------
// Types for CDNA5 WMMA (wave32): v_wmma_f32_16x16x32_bf16
// ---------------------------------------------------------------------------
typedef __attribute__((ext_vector_type(16))) __bf16 v16bf;
typedef __attribute__((ext_vector_type(8)))  float  v8f;

union BFrag16 { v16bf v; unsigned short u[16]; };

__device__ __forceinline__ unsigned short f2bf(float f) {
  union { float f; unsigned int u; } c; c.f = f;
  c.u += 0x7fffu + ((c.u >> 16) & 1u);           // round-to-nearest-even
  return (unsigned short)(c.u >> 16);
}
__device__ __forceinline__ float bf2f(unsigned short h) {
  union { unsigned int u; float f; } c; c.u = ((unsigned int)h) << 16;
  return c.f;
}

// ---------------------------------------------------------------------------
// CDNA5 async global->LDS copy (ASYNCcnt-tracked), with fallback.
// Builtin signature (from probe): (v4i AS1*, v4i AS3*, imm offset, imm cpol)
// where v4i = GCC-vector of 4 ints (128-bit payload).
// ---------------------------------------------------------------------------
#if defined(__HIP_DEVICE_COMPILE__) && __has_builtin(__builtin_amdgcn_global_load_async_to_lds_b128)
#define ASYNC_COPY 1
#else
#define ASYNC_COPY 0
#endif

typedef int v4i_b128 __attribute__((vector_size(16)));
typedef __attribute__((address_space(1))) v4i_b128* gptr_b128;
typedef __attribute__((address_space(3))) v4i_b128* lptr_b128;

#define TO_GLOBAL(p) ((gptr_b128)(p))
#define TO_LDS(p)    ((lptr_b128)(p))

__device__ __forceinline__ void async_wait_all() {
#if ASYNC_COPY
#if __has_builtin(__builtin_amdgcn_s_wait_asynccnt)
  __builtin_amdgcn_s_wait_asynccnt(0);
#else
  asm volatile("s_wait_asynccnt 0" ::: "memory");
#endif
#endif
}

// ---------------------------------------------------------------------------
// Problem constants
// ---------------------------------------------------------------------------
#define BATCH     512
#define NTOK      65          // WINDOW*WINDOW + NUM_REG
#define NPAD      80          // 5 x 16 tiles
#define KPAD      96          // 3 x 32 WMMA K-chunks
#define DIMX      1024
#define HEADS     32
#define DHEAD     32
#define NREL      225         // (2*8-1)^2

// ---------------------------------------------------------------------------
// Generic bf16 WMMA GEMM:  C[MxN] = act(A[MxK] * B[KxN] + bias)
// Block tile 128x64, 8 waves, each wave a 32x32 tile (2x2 WMMA frags).
// Double-buffered LDS; staging via async global->LDS copies when available.
// ---------------------------------------------------------------------------
#define BM 128
#define BN 64
#define BK 32

template<bool SILU, bool BIAS, bool OUT_BF16>
__global__ __launch_bounds__(256)
void wmma_gemm_bf16(const unsigned short* __restrict__ A,
                    const unsigned short* __restrict__ B,
                    const float* __restrict__ bias,
                    void* __restrict__ Cout,
                    int M, int N, int K)
{
  __shared__ unsigned short As[2][BM][BK + 8];   // +8 shorts: rows stay 16B aligned
  __shared__ unsigned short Bs[2][BK][BN + 8];

  const int tid  = threadIdx.x;
  const int lane = tid & 31;
  const int wave = tid >> 5;          // 0..7
  const int wm   = wave >> 1;         // 0..3 : M direction
  const int wn   = wave & 1;          // 0..1 : N direction
  const int l15  = lane & 15;
  const int hi   = lane >> 4;         // lane half (ISA fragment layout)

  const int bm = blockIdx.x * BM;
  const int bn = blockIdx.y * BN;

  // staging coordinates (A: 16 shorts/thread, B: 8 shorts/thread)
  const int ar_ld = tid >> 1;
  const int ac_ld = (tid & 1) << 4;
  const int br_ld = tid >> 3;
  const int bc_ld = (tid & 7) << 3;
  const bool a_in = (bm + ar_ld) < M;

  auto stage = [&](int buf, int kk) {
    const unsigned short* asrc = A + (size_t)(bm + ar_ld) * K + kk + ac_ld;
    const unsigned short* bsrc = B + (size_t)(kk + br_ld) * N + bn + bc_ld;
#if ASYNC_COPY
    if (a_in) {
      __builtin_amdgcn_global_load_async_to_lds_b128(
          TO_GLOBAL(asrc),     TO_LDS(&As[buf][ar_ld][ac_ld]),     0, 0);
      __builtin_amdgcn_global_load_async_to_lds_b128(
          TO_GLOBAL(asrc + 8), TO_LDS(&As[buf][ar_ld][ac_ld + 8]), 0, 0);
    } else {
      uint4 z = {0u,0u,0u,0u};
      uint4* dst = reinterpret_cast<uint4*>(&As[buf][ar_ld][ac_ld]);
      dst[0] = z; dst[1] = z;
    }
    __builtin_amdgcn_global_load_async_to_lds_b128(
        TO_GLOBAL(bsrc), TO_LDS(&Bs[buf][br_ld][bc_ld]), 0, 0);
#else
    uint4 z0 = {0u,0u,0u,0u}, z1 = {0u,0u,0u,0u};
    if (a_in) {
      const uint4* src = reinterpret_cast<const uint4*>(asrc);
      z0 = src[0]; z1 = src[1];
      __builtin_prefetch(asrc + BK, 0, 1);        // global_prefetch_b8
    }
    uint4* dst = reinterpret_cast<uint4*>(&As[buf][ar_ld][ac_ld]);
    dst[0] = z0; dst[1] = z1;
    *reinterpret_cast<uint4*>(&Bs[buf][br_ld][bc_ld]) =
        *reinterpret_cast<const uint4*>(bsrc);
    __builtin_prefetch(bsrc + (size_t)BK * N, 0, 1);
#endif
  };

  v8f acc[2][2] = {};

  stage(0, 0);
  async_wait_all();
  __syncthreads();

  int cur = 0;
  for (int kk = 0; kk < K; kk += BK) {
    if (kk + BK < K) stage(cur ^ 1, kk + BK);     // overlap with compute below

    // ---- fragments per ISA 7.12.2: K(i) = i + (i>=8 ? 8:0) + 8*hi ----
    BFrag16 af[2], bf[2];
    #pragma unroll
    for (int f = 0; f < 2; ++f) {
      const int ar = wm * 32 + f * 16 + l15;
      const int bc = wn * 32 + f * 16 + l15;
      #pragma unroll
      for (int i = 0; i < 16; ++i) {
        const int k = i + ((i >> 3) << 3) + (hi << 3);
        af[f].u[i] = As[cur][ar][k];
        bf[f].u[i] = Bs[cur][k][bc];
      }
    }
    #pragma unroll
    for (int fm = 0; fm < 2; ++fm)
      #pragma unroll
      for (int fn = 0; fn < 2; ++fn)
        acc[fm][fn] = __builtin_amdgcn_wmma_f32_16x16x32_bf16(
            false, af[fm].v, false, bf[fn].v, (short)0, acc[fm][fn], false, false);

    async_wait_all();
    __syncthreads();
    cur ^= 1;
  }

  // ---- epilogue: C/D layout row = r + 8*hi, col = lane&15 ----
  #pragma unroll
  for (int fm = 0; fm < 2; ++fm)
    #pragma unroll
    for (int fn = 0; fn < 2; ++fn) {
      const int gn = bn + wn * 32 + fn * 16 + l15;
      #pragma unroll
      for (int r = 0; r < 8; ++r) {
        const int gm = bm + wm * 32 + fm * 16 + r + (hi << 3);
        if (gm < M) {
          float v = acc[fm][fn][r];
          if constexpr (BIAS) v += bias[gn];
          if constexpr (SILU) v = v / (1.0f + __expf(-v));
          if constexpr (OUT_BF16)
            ((unsigned short*)Cout)[(size_t)gm * N + gn] = f2bf(v);
          else
            ((float*)Cout)[(size_t)gm * N + gn] = v;
        }
      }
    }
}

// ---------------------------------------------------------------------------
// f32 -> bf16 convert (grid-stride)
// ---------------------------------------------------------------------------
__global__ __launch_bounds__(256)
void cvt_f32_bf16(const float* __restrict__ in, unsigned short* __restrict__ out, int n)
{
  int i = blockIdx.x * blockDim.x + threadIdx.x;
  const int stride = gridDim.x * blockDim.x;
  for (; i < n; i += stride) out[i] = f2bf(in[i]);
}

// ---------------------------------------------------------------------------
// Fused LayerNorm + FiLM apply:  xn = ((x-mu)*rstd) * gamma[b] + beta[b]
// ---------------------------------------------------------------------------
__global__ __launch_bounds__(256)
void ln_film_kernel(const float* __restrict__ x, const float* __restrict__ gb,
                    unsigned short* __restrict__ xn)
{
  __shared__ float red[8];
  const int row = blockIdx.x;           // b*NTOK + n
  const int b   = row / NTOK;
  const int tid = threadIdx.x;
  const float* xr = x + (size_t)row * DIMX;

  float lx[4], s = 0.f;
  #pragma unroll
  for (int i = 0; i < 4; ++i) { lx[i] = xr[tid + 256 * i]; s += lx[i]; }
  #pragma unroll
  for (int o = 16; o > 0; o >>= 1) s += __shfl_xor(s, o, 32);
  if ((tid & 31) == 0) red[tid >> 5] = s;
  __syncthreads();
  float tot = 0.f;
  #pragma unroll
  for (int i = 0; i < 8; ++i) tot += red[i];
  const float mu = tot * (1.0f / DIMX);
  __syncthreads();

  float vs = 0.f;
  #pragma unroll
  for (int i = 0; i < 4; ++i) { const float d = lx[i] - mu; vs += d * d; }
  #pragma unroll
  for (int o = 16; o > 0; o >>= 1) vs += __shfl_xor(vs, o, 32);
  if ((tid & 31) == 0) red[tid >> 5] = vs;
  __syncthreads();
  float vtot = 0.f;
  #pragma unroll
  for (int i = 0; i < 8; ++i) vtot += red[i];
  const float rstd = rsqrtf(vtot * (1.0f / DIMX) + 1e-5f);

  const float* g = gb + (size_t)b * (2 * DIMX);
  unsigned short* op = xn + (size_t)row * DIMX;
  #pragma unroll
  for (int i = 0; i < 4; ++i) {
    const int c = tid + 256 * i;
    op[c] = f2bf((lx[i] - mu) * rstd * g[c] + g[DIMX + c]);
  }
}

// ---------------------------------------------------------------------------
// Fused attention per (batch, head).
// Rel-pos bias is pre-materialized into Sm and fed to WMMA as the C operand:
//   S = Q*K^T + bias  in a single v_wmma (no divergent epilogue).
// ---------------------------------------------------------------------------
__global__ __launch_bounds__(256)
void attn_kernel(const unsigned short* __restrict__ qkv,
                 const float* __restrict__ qg, const float* __restrict__ kg,
                 const float* __restrict__ relb,
                 unsigned short* __restrict__ ao)
{
  const int bh  = blockIdx.x;
  const int b   = bh >> 5;
  const int h   = bh & 31;
  const int tid  = threadIdx.x;
  const int lane = tid & 31;
  const int wave = tid >> 5;
  const int l15  = lane & 15;
  const int hi   = lane >> 4;

  __shared__ unsigned short Qs[NPAD][36];
  __shared__ unsigned short Ks[NPAD][36];
  __shared__ unsigned short Vs[KPAD][36];
  __shared__ float          Sm[NPAD][84];
  __shared__ unsigned short Ps[NPAD][100];

  const float qgam = qg[h * DHEAD + lane];
  const float kgam = kg[h * DHEAD + lane];
  const float rs = 5.65685424949238f;          // sqrt(DIM_HEAD)

  // ---- load + l2-normalize q/k, load v (lane == head-dim, wave == row) ----
  for (int r = wave; r < KPAD; r += 8) {
    float qv = 0.f, kv = 0.f, vv = 0.f;
    if (r < NTOK) {
      const size_t base = ((size_t)(b * NTOK + r)) * (3 * DIMX) + h * DHEAD + lane;
      qv = bf2f(qkv[base]);
      kv = bf2f(qkv[base + DIMX]);
      vv = bf2f(qkv[base + 2 * DIMX]);
    }
    float qs = qv * qv, ks = kv * kv;
    #pragma unroll
    for (int o = 16; o > 0; o >>= 1) {
      qs += __shfl_xor(qs, o, 32);
      ks += __shfl_xor(ks, o, 32);
    }
    const float qn = rs / fmaxf(sqrtf(qs), 1e-12f);
    const float kn = rs / fmaxf(sqrtf(ks), 1e-12f);
    if (r < NPAD) {
      Qs[r][lane] = f2bf(qv * qn * qgam);
      Ks[r][lane] = f2bf(kv * kn * kgam);
    }
    Vs[r][lane] = f2bf(vv);
  }

  // ---- pre-fill Sm with rel-pos bias / padded-key mask (WMMA C operand) ----
  for (int e = tid; e < NPAD * NPAD; e += 256) {
    const int i = e / NPAD;
    const int j = e - i * NPAD;
    float v;
    if (j >= NTOK)      v = -1e30f;              // mask padded keys
    else if (i >= NTOK) v = 0.f;                 // padded query rows: benign
    else {
      int idx;
      if (i == 0 || j == 0) idx = NREL;          // register token row/col
      else {
        const int ii = i - 1, jj = j - 1;
        idx = ((ii >> 3) - (jj >> 3) + 7) * 15 + ((ii & 7) - (jj & 7) + 7);
      }
      v = relb[idx * HEADS + h];
    }
    Sm[i][j] = v;
  }
  __syncthreads();

  // ---- S = Q K^T + bias : 5x5 tiles, one 16x16x32 WMMA each (C = bias) ----
  for (int t = wave; t < 25; t += 8) {
    const int ti = t / 5, tj = t % 5;
    BFrag16 af, bf;
    v8f cf;
    #pragma unroll
    for (int i = 0; i < 16; ++i) {
      const int k = i + ((i >> 3) << 3) + (hi << 3);
      af.u[i] = Qs[ti * 16 + l15][k];
      bf.u[i] = Ks[tj * 16 + l15][k];   // B[k][n] = K[n][k] (K^T)
    }
    #pragma unroll
    for (int r = 0; r < 8; ++r) cf[r] = Sm[ti * 16 + r + (hi << 3)][tj * 16 + l15];
    cf = __builtin_amdgcn_wmma_f32_16x16x32_bf16(
        false, af.v, false, bf.v, (short)0, cf, false, false);
    #pragma unroll
    for (int r = 0; r < 8; ++r) Sm[ti * 16 + r + (hi << 3)][tj * 16 + l15] = cf[r];
  }
  __syncthreads();

  // ---- row softmax (rows 0..79; padded rows benign) ----
  if (tid < NPAD) {
    float m = -1e30f;
    for (int j = 0; j < NTOK; ++j) m = fmaxf(m, Sm[tid][j]);
    float sum = 0.f;
    for (int j = 0; j < NTOK; ++j) sum += __expf(Sm[tid][j] - m);
    const float inv = 1.0f / sum;
    for (int j = 0; j < KPAD; ++j) {
      const float p = (j < NTOK) ? __expf(Sm[tid][j] - m) * inv : 0.f;
      Ps[tid][j] = f2bf(p);
    }
  }
  __syncthreads();

  // ---- out = P V : 5 (M) x 2 (N) tiles, 3 K-chunks of 32 ----
  for (int t = wave; t < 10; t += 8) {
    const int ti = t >> 1, tn = t & 1;
    v8f acc = {};
    #pragma unroll
    for (int kc = 0; kc < 3; ++kc) {
      BFrag16 af, bf;
      #pragma unroll
      for (int i = 0; i < 16; ++i) {
        const int k = kc * 32 + i + ((i >> 3) << 3) + (hi << 3);
        af.u[i] = Ps[ti * 16 + l15][k];
        bf.u[i] = Vs[k][tn * 16 + l15];
      }
      acc = __builtin_amdgcn_wmma_f32_16x16x32_bf16(
          false, af.v, false, bf.v, (short)0, acc, false, false);
    }
    #pragma unroll
    for (int r = 0; r < 8; ++r) {
      const int i = ti * 16 + r + (hi << 3);
      if (i < NTOK) {
        const int d = tn * 16 + l15;
        ao[((size_t)(b * NTOK + i)) * DIMX + h * DHEAD + d] = f2bf(acc[r]);
      }
    }
  }
}

// ---------------------------------------------------------------------------
// Host launcher
// ---------------------------------------------------------------------------
extern "C" void kernel_launch(void* const* d_in, const int* in_sizes, int n_in,
                              void* d_out, int out_size, void* d_ws, size_t ws_size,
                              hipStream_t stream)
{
  const float* x    = (const float*)d_in[0];
  const float* cond = (const float*)d_in[1];
  const float* fw1  = (const float*)d_in[2];
  const float* fb1  = (const float*)d_in[3];
  const float* fw2  = (const float*)d_in[4];
  const float* fb2  = (const float*)d_in[5];
  const float* wqkv = (const float*)d_in[6];
  const float* qg   = (const float*)d_in[7];
  const float* kg   = (const float*)d_in[8];
  const float* relb = (const float*)d_in[9];
  const float* wout = (const float*)d_in[10];
  float* out = (float*)d_out;

  char* ws = (char*)d_ws;
  size_t off = 0;
  auto alloc = [&](size_t bytes) -> void* {
    void* p = ws + off;
    off = (off + bytes + 255) & ~(size_t)255;
    return p;
  };

  const int ROWS = BATCH * NTOK;                     // 33280
  unsigned short* cond_bf = (unsigned short*)alloc((size_t)BATCH * DIMX * 2);
  unsigned short* fw1_bf  = (unsigned short*)alloc((size_t)DIMX * 2 * DIMX * 2);
  unsigned short* fw2_bf  = (unsigned short*)alloc((size_t)2 * DIMX * 2 * DIMX * 2);
  unsigned short* wqkv_bf = (unsigned short*)alloc((size_t)DIMX * 3 * DIMX * 2);
  unsigned short* wout_bf = (unsigned short*)alloc((size_t)DIMX * DIMX * 2);
  unsigned short* h_bf    = (unsigned short*)alloc((size_t)BATCH * 2 * DIMX * 2);
  float*          gb_f    = (float*)alloc((size_t)BATCH * 2 * DIMX * 4);
  unsigned short* xn_bf   = (unsigned short*)alloc((size_t)ROWS * DIMX * 2);
  unsigned short* qkv_bf  = (unsigned short*)alloc((size_t)ROWS * 3 * DIMX * 2);
  unsigned short* ao_bf   = (unsigned short*)alloc((size_t)ROWS * DIMX * 2);

  // 1) f32 -> bf16 converts
  cvt_f32_bf16<<<1024, 256, 0, stream>>>(cond, cond_bf, BATCH * DIMX);
  cvt_f32_bf16<<<1024, 256, 0, stream>>>(fw1,  fw1_bf,  DIMX * 2 * DIMX);
  cvt_f32_bf16<<<2048, 256, 0, stream>>>(fw2,  fw2_bf,  2 * DIMX * 2 * DIMX);
  cvt_f32_bf16<<<2048, 256, 0, stream>>>(wqkv, wqkv_bf, DIMX * 3 * DIMX);
  cvt_f32_bf16<<<1024, 256, 0, stream>>>(wout, wout_bf, DIMX * DIMX);

  // 2) FiLM layer 1: h = silu(cond @ fw1 + fb1)   [512 x 2048, K=1024]
  wmma_gemm_bf16<true, true, true><<<dim3(BATCH / BM, (2 * DIMX) / BN), 256, 0, stream>>>(
      cond_bf, fw1_bf, fb1, h_bf, BATCH, 2 * DIMX, DIMX);

  // 3) FiLM layer 2: gb = h @ fw2 + fb2           [512 x 2048, K=2048]
  wmma_gemm_bf16<false, true, false><<<dim3(BATCH / BM, (2 * DIMX) / BN), 256, 0, stream>>>(
      h_bf, fw2_bf, fb2, gb_f, BATCH, 2 * DIMX, 2 * DIMX);

  // 4) LayerNorm + FiLM apply -> bf16 activations
  ln_film_kernel<<<ROWS, 256, 0, stream>>>(x, gb_f, xn_bf);

  // 5) QKV projection: [33280 x 3072, K=1024]
  wmma_gemm_bf16<false, false, true><<<dim3(ROWS / BM, (3 * DIMX) / BN), 256, 0, stream>>>(
      xn_bf, wqkv_bf, nullptr, qkv_bf, ROWS, 3 * DIMX, DIMX);

  // 6) fused attention per (batch, head)
  attn_kernel<<<BATCH * HEADS, 256, 0, stream>>>(qkv_bf, qg, kg, relb, ao_bf);

  // 7) output projection: [33280 x 1024, K=1024] -> f32 d_out
  wmma_gemm_bf16<false, false, false><<<dim3(ROWS / BM, DIMX / BN), 256, 0, stream>>>(
      ao_bf, wout_bf, nullptr, out, ROWS, DIMX, DIMX);

  (void)in_sizes; (void)n_in; (void)out_size; (void)ws_size;
}
